// Net_86689619902655
// MI455X (gfx1250) — matrix operand order repaired and checked by reading, not compiled, not run on hardware
//
#include <hip/hip_runtime.h>
#include <hip/hip_bf16.h>

typedef __attribute__((ext_vector_type(16))) _Float16 v16h;
typedef __attribute__((ext_vector_type(8)))  float    v8f;
typedef __attribute__((ext_vector_type(4)))  unsigned int v4u;
typedef __attribute__((ext_vector_type(8)))  int      v8i;
typedef __attribute__((ext_vector_type(4)))  int      v4i;

#define LEAKY_ALPHA 0.1f
#define BN_EPS 1e-5f

#if __has_builtin(__builtin_amdgcn_tensor_load_to_lds)
#define USE_TDM 1
#else
#define USE_TDM 0
#endif

union V16 { v16h v; uint4 u[2]; };
union H8  { _Float16 h[8]; uint4 u; };

#if USE_TDM
// TDM: DMA a 64(n) x 32(k) f16 tile (row stride KpadHalves in global) into LDS,
// inserting 16B of LDS padding after every 64B row => LDS row stride 80B (40 halves),
// matching the WMMA fragment loads. D# packed per CDNA5 ISA 8.3-8.6.
// clang-23 toolchain: 6-arg form (v4u, v8i, v4i, v4i, v8i, i32 cpol).
static __device__ __forceinline__ void tdm_load_b_tile(const _Float16* gsrc,
                                                       unsigned ldsOff,
                                                       long KpadHalves)
{
    unsigned long long ga = (unsigned long long)(uintptr_t)gsrc;
    v4u g0;
    g0[0] = 1u;                                        // count=1, user mode
    g0[1] = ldsOff;                                    // lds_addr (bytes)
    g0[2] = (unsigned)(ga & 0xFFFFFFFFull);            // global_addr[31:0]
    g0[3] = (unsigned)((ga >> 32) & 0x1FFFFFFull)      // global_addr[56:32]
          | (2u << 30);                                // type = 2 (image)

    unsigned w[8] = {0,0,0,0,0,0,0,0};
    w[0] = (1u << 16)      // data_size = 2 bytes
         | (1u << 20)      // pad_enable
         | (3u << 22)      // pad_interval: 16 DWORDs (64B) between pads
         | (3u << 25);     // pad_amount: 4 DWORDs (16B)
    const unsigned td0 = 32, td1 = 64;                 // tensor dims == tile dims: no OOB
    w[1] |= (td0 & 0xFFFFu) << 16;                     // tensor_dim0 [79:48]
    w[2] |= td0 >> 16;
    w[2] |= (td1 & 0xFFFFu) << 16;                     // tensor_dim1 [111:80]
    w[3] |= td1 >> 16;
    w[3] |= 32u << 16;                                 // tile_dim0 = 32 elements
    w[4]  = 64u;                                       // tile_dim1 = 64, tile_dim2 = 0
    unsigned long long s0 = (unsigned long long)KpadHalves;
    w[5]  = (unsigned)(s0 & 0xFFFFFFFFull);            // tensor_dim0_stride [207:160]
    w[6] |= (unsigned)((s0 >> 32) & 0xFFFFull);

    v8i g1;
    #pragma unroll
    for (int i = 0; i < 8; ++i) g1[i] = (int)w[i];
    v4i g2 = {0, 0, 0, 0};
    v4i g3 = {0, 0, 0, 0};
    v8i g4 = {0, 0, 0, 0, 0, 0, 0, 0};
    __builtin_amdgcn_tensor_load_to_lds(g0, g1, g2, g3, g4, 0);
}
#endif

// Implicit-GEMM conv. M = B*H*W, N = CoutPad (64-aligned), K loop over Kpad (32-aligned).
// Block: 256 threads (8 waves). Block tile 256(M) x 64(N); wave w owns rows 32w..32w+31
// -> 2 A-frags x 4 B-frags = 8 v_wmma_f32_16x16x32_f16 per 32-wide K step.
// Weights pre-packed f16 [CoutPad][Kpad] (zero padded); B tile arrives via TDM.
// mode 0: fused BN+LeakyReLU, NCHW f32 store. mode 1: final 1x1 conv, one image per
// block (M tile == HW == 256): bias+leaky+spatial-mean+gate, LDS-reduced, out +=.
__global__ __launch_bounds__(256)
void conv_wmma_kernel(const float* __restrict__ in, const _Float16* __restrict__ wp,
                      const float* __restrict__ p0, const float* __restrict__ p1,
                      const float* __restrict__ p2, const float* __restrict__ p3,
                      float* __restrict__ out, const float* __restrict__ gates,
                      int Cin, int Cout, int H, int W, int Ktot, int Kpad,
                      int k3, int mode, int expert)
{
    __shared__ alignas(16) _Float16 sA[256 * 40];   // 20.0 KB, row stride 80B
    __shared__ alignas(16) _Float16 sB[64 * 40];    //  5.0 KB
    __shared__ float sRed[64 * 16];                 //  4.0 KB (mode-1 reduction)

    const int t    = threadIdx.x;
    const int lane = t & 31;
    const int wave = t >> 5;
    const int HW   = H * W;
    const int mBase = blockIdx.x * 256;
    const int nBase = blockIdx.y * 64;

    // A staging: thread t owns LDS row t (one im2col row, 32 halves per K step)
    const int mA   = mBase + t;
    const int bImg = mA / HW;
    const int pA   = mA - bImg * HW;
    const int hA   = pA / W;
    const int wA   = pA - hA * W;

    v8f acc[2][4];
#pragma unroll
    for (int g = 0; g < 2; ++g)
#pragma unroll
        for (int s = 0; s < 4; ++s)
#pragma unroll
            for (int v = 0; v < 8; ++v) acc[g][s][v] = 0.0f;

    for (int kt = 0; kt < Kpad; kt += 32) {
#if USE_TDM
        if (wave == 0)   // DMA B tile: overlaps with the A-gather below
            tdm_load_b_tile(wp + (long)nBase * Kpad + kt,
                            (unsigned)(uintptr_t)(void*)sB, (long)Kpad);
#else
        if (t < 128) {   // manual B staging: 16 contiguous halves per thread
            int r = t >> 1, kb = (t & 1) * 16;
            const uint4* src = (const uint4*)&wp[(long)(nBase + r) * Kpad + kt + kb];
            *(uint4*)&sB[r * 40 + kb]     = src[0];
            *(uint4*)&sB[r * 40 + kb + 8] = src[1];
        }
#endif
        // ---- stage A tile (256 x 32 f16): im2col gather + f32->f16 ----
        {
            H8 grp;
#pragma unroll
            for (int i = 0; i < 32; ++i) {
                int k = kt + i;
                float val = 0.0f;
                if (k < Ktot) {
                    if (k3) {
                        int ci = k / 9, r9 = k - ci * 9;
                        int kh = r9 / 3, kw = r9 - kh * 3;
                        int ih = hA + kh - 1, iw = wA + kw - 1;
                        if ((unsigned)ih < (unsigned)H && (unsigned)iw < (unsigned)W)
                            val = in[((long)bImg * Cin + ci) * HW + ih * W + iw];
                    } else {
                        val = in[((long)bImg * Cin + k) * HW + pA];
                    }
                }
                grp.h[i & 7] = (_Float16)val;
                if ((i & 7) == 7) *(uint4*)&sA[t * 40 + (i - 7)] = grp.u;
            }
        }
#if USE_TDM
        if (wave == 0) __builtin_amdgcn_s_wait_tensorcnt(0);
#endif
        __syncthreads();

        // ---- A fragments (ISA 7.12.2: lanes 0-15 K0-7/16-23, lanes 16-31 K8-15/24-31)
        V16 a0, a1;
        {
            int mr0 = wave * 32 + (lane & 15);
            int ko  = (lane < 16) ? 0 : 8;
            a0.u[0] = *(const uint4*)&sA[mr0 * 40 + ko];
            a0.u[1] = *(const uint4*)&sA[mr0 * 40 + ko + 16];
            a1.u[0] = *(const uint4*)&sA[(mr0 + 16) * 40 + ko];
            a1.u[1] = *(const uint4*)&sA[(mr0 + 16) * 40 + ko + 16];
        }
#pragma unroll
        for (int s = 0; s < 4; ++s) {
            V16 bf;                         // lane = column n; lanes 0-15 K0-7/16-23, 16-31 K8-15/24-31
            int nr = s * 16 + (lane & 15);
            int ko = (lane < 16) ? 0 : 8;
            bf.u[0] = *(const uint4*)&sB[nr * 40 + ko];
            bf.u[1] = *(const uint4*)&sB[nr * 40 + ko + 16];
            acc[0][s] = __builtin_amdgcn_wmma_f32_16x16x32_f16(
                false, a0.v, false, bf.v, (short)0, acc[0][s], false, false);
            acc[1][s] = __builtin_amdgcn_wmma_f32_16x16x32_f16(
                false, a1.v, false, bf.v, (short)0, acc[1][s], false, false);
        }
        __syncthreads();
    }

    // D layout per 16x16 subtile: element (m = v + 8*(lane>>4), n = lane&15)
    const int rOff = (lane >> 4) * 8;
    if (mode == 0) {
#pragma unroll
        for (int s = 0; s < 4; ++s) {
            int n = nBase + s * 16 + (lane & 15);
            if (n >= Cout) continue;
            float sc = p0[n] * rsqrtf(p3[n] + BN_EPS);   // gamma / sqrt(var+eps)
            float sh = p1[n] - p2[n] * sc;               // beta - mean*scale
#pragma unroll
            for (int g = 0; g < 2; ++g) {
#pragma unroll
                for (int v = 0; v < 8; ++v) {
                    int m = mBase + wave * 32 + g * 16 + rOff + v;
                    int bi = m / HW;
                    int p  = m - bi * HW;
                    float val = acc[g][s][v] * sc + sh;
                    val = (val >= 0.0f) ? val : LEAKY_ALPHA * val;
                    out[((long)bi * Cout + n) * HW + p] = val;
                }
            }
        }
    } else {
        // mode 1: block == one image (HW == 256). Deterministic LDS reduction.
        int bi = mBase / HW;
#pragma unroll
        for (int s = 0; s < 4; ++s) {
            int n = nBase + s * 16 + (lane & 15);
            float bias = (n < Cout) ? p1[n] : 0.0f;
            float sum = 0.0f;
#pragma unroll
            for (int g = 0; g < 2; ++g)
#pragma unroll
                for (int v = 0; v < 8; ++v) {
                    float val = acc[g][s][v] + bias;
                    sum += (val >= 0.0f) ? val : LEAKY_ALPHA * val;
                }
            // column slot: 16 contributors = 8 waves x 2 lane-halves
            sRed[(s * 16 + (lane & 15)) * 16 + wave * 2 + (lane >> 4)] = sum;
        }
        __syncthreads();
        if (t < 64) {
            int n = nBase + t;
            if (n < Cout) {
                float tot = 0.0f;
#pragma unroll
                for (int j = 0; j < 16; ++j) tot += sRed[t * 16 + j];
                float gw = gates[bi * 8 + expert] * (1.0f / 256.0f);
                out[(long)bi * 1000 + n] += tot * gw;   // experts serialized on stream
            }
        }
    }
}

// Pack f32 OIHW weights -> f16 [CoutPad][Kpad], k = ci*kk + kh*k + kw, zero padded.
__global__ void pack_w_kernel(const float* __restrict__ w, _Float16* __restrict__ wp,
                              int Cout, int K, int CoutPad, int Kpad)
{
    int idx = blockIdx.x * blockDim.x + threadIdx.x;
    int total = CoutPad * Kpad;
    if (idx >= total) return;
    int co = idx / Kpad;
    int k  = idx - co * Kpad;
    float val = (co < Cout && k < K) ? w[(long)co * K + k] : 0.0f;
    wp[idx] = (_Float16)val;
}

__global__ void maxpool2_kernel(const float* __restrict__ in, float* __restrict__ out,
                                int C, int H, int W, int total)
{
    int idx = blockIdx.x * blockDim.x + threadIdx.x;
    if (idx >= total) return;
    int W2 = W >> 1, H2 = H >> 1;
    int w2 = idx % W2; int tmp = idx / W2;
    int h2 = tmp % H2; tmp /= H2;
    int c  = tmp % C;  int b = tmp / C;
    const float* base = in + ((long)b * C + c) * H * W + (h2 * 2) * W + w2 * 2;
    float m0 = fmaxf(base[0], base[1]);
    float m1 = fmaxf(base[W], base[W + 1]);
    out[idx] = fmaxf(m0, m1);
}

__global__ __launch_bounds__(64)
void gate_kernel(const float* __restrict__ x,
                 const float* __restrict__ w1, const float* __restrict__ b1,
                 const float* __restrict__ w2, const float* __restrict__ b2,
                 float* __restrict__ gates)
{
    __shared__ float gi[48];
    __shared__ float hid[64];
    __shared__ float lg[8];
    int b = blockIdx.x, t = threadIdx.x;
    if (t < 48) {                       // adaptive avg pool x -> (3,4,4)
        int c = t / 16, r = t % 16;
        int ph = r / 4, pw = r % 4;
        const float* px = x + ((long)b * 3 + c) * 4096 + ph * 16 * 64 + pw * 16;
        float s = 0.0f;
        for (int i = 0; i < 16; ++i)
            for (int j = 0; j < 16; ++j) s += px[i * 64 + j];
        gi[t] = s * (1.0f / 256.0f);
    }
    __syncthreads();
    {
        float h = b1[t];
        for (int i = 0; i < 48; ++i) h += gi[i] * w1[i * 64 + t];
        hid[t] = fmaxf(h, 0.0f);
    }
    __syncthreads();
    if (t < 8) {
        float l = b2[t];
        for (int j = 0; j < 64; ++j) l += hid[j] * w2[j * 8 + t];
        lg[t] = l;
    }
    __syncthreads();
    if (t == 0) {                       // top-2 (first-occurrence ties) + softmax
        int i0 = 0; float v0 = lg[0];
        for (int i = 1; i < 8; ++i) if (lg[i] > v0) { v0 = lg[i]; i0 = i; }
        int i1 = -1; float v1 = -3.4e38f;
        for (int i = 0; i < 8; ++i) if (i != i0 && lg[i] > v1) { v1 = lg[i]; i1 = i; }
        float mx = fmaxf(v0, v1);
        float e0 = __expf(v0 - mx), e1 = __expf(v1 - mx);
        float inv = 1.0f / (e0 + e1);
        for (int i = 0; i < 8; ++i) gates[b * 8 + i] = 0.0f;
        gates[b * 8 + i0] = e0 * inv;
        gates[b * 8 + i1] = e1 * inv;
    }
}

__global__ void zero_kernel(float* __restrict__ p, int n)
{
    int i = blockIdx.x * blockDim.x + threadIdx.x;
    if (i < n) p[i] = 0.0f;
}

extern "C" void kernel_launch(void* const* d_in, const int* in_sizes, int n_in,
                              void* d_out, int out_size, void* d_ws, size_t ws_size,
                              hipStream_t stream)
{
    static const int LEN[8][3] = {
        {2,2,2},{2,2,2},{2,2,2},{3,2,1},{2,2,2},{1,3,2},{2,2,2},{2,3,1}};
    static const int COUT[8][3][3] = {
        {{ 32, 32,  0},{ 64, 64,  0},{128,128,  0}},
        {{ 64, 64,  0},{128,128,  0},{256,256,  0}},
        {{ 48, 48,  0},{ 96, 96,  0},{192,192,  0}},
        {{ 64, 64, 64},{128,128,  0},{256,  0,  0}},
        {{ 96, 96,  0},{192,192,  0},{384,384,  0}},
        {{ 64,  0,  0},{128,128,128},{256,256,  0}},
        {{ 80, 80,  0},{160,160,  0},{320,320,  0}},
        {{ 64, 64,  0},{128,128,128},{256,  0,  0}},
    };

    const float* x = (const float*)d_in[0];
    float* out = (float*)d_out;
    char* ws = (char*)d_ws;

    float* gates = (float*)ws;                                          // 32*8 f32
    const size_t bufBytes = (size_t)32 * 96 * 64 * 64 * sizeof(float);  // 50.3 MB
    float*     bufA  = (float*)(ws + 1024);
    float*     bufB  = (float*)(ws + 1024 + bufBytes);
    _Float16*  wpack = (_Float16*)(ws + 1024 + 2 * bufBytes);           // <= 4 MB

    // gate tensors come after all expert tensors (each expert: 5 per unit + out_w + out_b)
    int gidx = 1;
    for (int e = 0; e < 8; ++e) {
        int u = 0;
        for (int s = 0; s < 3; ++s) u += LEN[e][s];
        gidx += 5 * u + 2;
    }
    const float* gw1 = (const float*)d_in[gidx + 0];
    const float* gb1 = (const float*)d_in[gidx + 1];
    const float* gw2 = (const float*)d_in[gidx + 2];
    const float* gb2 = (const float*)d_in[gidx + 3];

    zero_kernel<<<dim3((out_size + 255) / 256), dim3(256), 0, stream>>>(out, out_size);
    gate_kernel<<<dim3(32), dim3(64), 0, stream>>>(x, gw1, gb1, gw2, gb2, gates);

    int idx = 1;
    for (int e = 0; e < 8; ++e) {
        const float* cur = x;
        float* nxt = bufA;
        int cin = 3, H = 64, W = 64;
        for (int s = 0; s < 3; ++s) {
            int ln = LEN[e][s];
            for (int j = 0; j < ln; ++j) {
                int cout = COUT[e][s][j];
                int is3  = !(ln > 1 && (j % 2 == 0));   // pointwise if multi-unit stage, even j
                int K    = is3 ? cin * 9 : cin;
                int Kpad = (K + 31) & ~31;
                int CoutPad = (cout + 63) & ~63;
                const float* w  = (const float*)d_in[idx + 0];
                const float* ga = (const float*)d_in[idx + 1];
                const float* be = (const float*)d_in[idx + 2];
                const float* mn = (const float*)d_in[idx + 3];
                const float* vr = (const float*)d_in[idx + 4];
                idx += 5;
                int pk = CoutPad * Kpad;
                pack_w_kernel<<<dim3((pk + 255) / 256), dim3(256), 0, stream>>>(
                    w, wpack, cout, K, CoutPad, Kpad);
                int M = 32 * H * W;
                dim3 grid(M / 256, CoutPad / 64);
                conv_wmma_kernel<<<grid, dim3(256), 0, stream>>>(
                    cur, wpack, ga, be, mn, vr, nxt, gates,
                    cin, cout, H, W, K, Kpad, is3, /*mode=*/0, e);
                cur = nxt;
                nxt = (nxt == bufA) ? bufB : bufA;
                cin = cout;
            }
            if (s != 2) {
                int H2 = H / 2, W2 = W / 2;
                int total = 32 * cin * H2 * W2;
                maxpool2_kernel<<<dim3((total + 255) / 256), dim3(256), 0, stream>>>(
                    cur, nxt, cin, H, W, total);
                cur = nxt;
                nxt = (nxt == bufA) ? bufB : bufA;
                H = H2; W = W2;
            }
        }
        // final 1x1 conv (cin -> 1000): one image per block, gate-weighted accumulate
        const float* ow = (const float*)d_in[idx + 0];
        const float* ob = (const float*)d_in[idx + 1];
        idx += 2;
        int K = cin, Kpad = (K + 31) & ~31, CoutPad = (1000 + 63) & ~63;  // 1024
        int pk = CoutPad * Kpad;
        pack_w_kernel<<<dim3((pk + 255) / 256), dim3(256), 0, stream>>>(
            ow, wpack, 1000, K, CoutPad, Kpad);
        int M = 32 * H * W;   // 8192
        dim3 grid(M / 256, CoutPad / 64);
        conv_wmma_kernel<<<grid, dim3(256), 0, stream>>>(
            cur, wpack, ob, ob, ob, ob, out, gates,
            cin, 1000, H, W, K, Kpad, /*k3=*/0, /*mode=*/1, e);
    }
    (void)in_sizes; (void)n_in; (void)ws_size;
}
